// MultiheadAttention_15547781611527
// MI455X (gfx1250) — matrix operand order repaired and checked by reading, not compiled, not run on hardware
//
#include <hip/hip_runtime.h>
#include <hip/hip_bf16.h>
#include <math.h>
#include <stdint.h>

// ---------------------------------------------------------------------------
// MI455X (gfx1250) multi-head attention forward, wave32 + WMMA bf16 16x16x32.
// HBM-bound on the B*H*S*S attn tensor. GEMMs double-buffer LDS tiles; bf16
// sources are staged with GLOBAL_LOAD_ASYNC_TO_LDS_B128 (ASYNCcnt), fp32
// sources go through v_cvt_pk_bf16_f32 + ds_store_b128.
// ---------------------------------------------------------------------------

typedef __attribute__((ext_vector_type(16))) __bf16 v16bf;
typedef __attribute__((ext_vector_type(8)))  __bf16 v8bf;
typedef __attribute__((ext_vector_type(8)))  float  v8f;

#define D_MODEL   1024
#define NUM_HEADS 16
#define DEPTH     64
#define BATCH     8
#define SEQ       1024
#define BS        (BATCH * SEQ)        // 8192 tokens
#define BH        (BATCH * NUM_HEADS)  // 128 (b,h) pairs

// Block tile 128x64, K-step 32; 8 waves in 4x2 grid, wave tile 32x32 (2x2 frags)
#define BM 128
#define BN 64
#define BK 32
#define THREADS 256

// --- async global -> LDS copy (CDNA5, GV mode, 16 bytes per lane) -----------
// Generic pointers to __shared__ carry the LDS byte offset in addr[31:0]
// (ISA 10.2 aperture mapping), which is exactly the VDST operand.
__device__ __forceinline__ void async_ld16(void* ldsDst, const void* gsrc) {
  uint32_t loff = (uint32_t)(uintptr_t)ldsDst;
  asm volatile("global_load_async_to_lds_b128 %0, %1, off"
               :: "v"(loff), "v"(gsrc)
               : "memory");
}
__device__ __forceinline__ void async_wait0() {
  asm volatile("s_wait_asynccnt 0x0" ::: "memory");
}

// --- fragment-major LDS offsets ---------------------------------------------
// A subtile 16x32 bf16 (ISA 7.12.2): lane l: row m=l%16;
//   l<16 -> K {0..7,16..23}, l>=16 -> K {8..15,24..31}; 512 bf16 per subtile.
// 8 consecutive K of one row land in 8 consecutive LDS bf16 (one 16B xfer):
__device__ __forceinline__ int lds_off_A8(int m, int kg) {  // kg = k/8 in [0,4)
  int lane = ((kg & 1) << 4) + (m & 15);
  return (m >> 4) * 512 + lane * 16 + ((kg & 2) ? 8 : 0);
}
// B subtile 32x16 bf16: lane l: col n=l%16; l<16 -> K {0..7,16..23}, l>=16 -> K {8..15,24..31}
__device__ __forceinline__ int lds_off_B8(int n, int kq) {  // kq = k/8 in [0,4)
  int lane = ((kq & 1) << 4) + (n & 15);
  return (n >> 4) * 512 + lane * 16 + ((kq & 2) ? 8 : 0);
}

// --- staging: one task = 8 K-contiguous elements ----------------------------
__device__ __forceinline__ v8bf cvt8(const float* __restrict__ p) {
  float4 lo = *(const float4*)p;
  float4 hi = *(const float4*)(p + 4);
  v8bf o;
  o[0] = (__bf16)lo.x; o[1] = (__bf16)lo.y; o[2] = (__bf16)lo.z; o[3] = (__bf16)lo.w;
  o[4] = (__bf16)hi.x; o[5] = (__bf16)hi.y; o[6] = (__bf16)hi.z; o[7] = (__bf16)hi.w;
  return o;
}
// A tile: 128 rows x 4 kgroups = 512 tasks, 2 per thread.
__device__ __forceinline__ void stage_A_f32(__bf16* dst, const float* __restrict__ src,
                                            int ld, int tid) {
#pragma unroll
  for (int t = 0; t < 2; ++t) {
    int task = tid + t * THREADS;
    int m = task >> 2, kg = task & 3;
    *(v8bf*)(dst + lds_off_A8(m, kg)) = cvt8(src + (size_t)m * ld + kg * 8);
  }
}
__device__ __forceinline__ void stage_A_bf16(__bf16* dst, const __bf16* __restrict__ src,
                                             int ld, int tid) {
#pragma unroll
  for (int t = 0; t < 2; ++t) {
    int task = tid + t * THREADS;
    int m = task >> 2, kg = task & 3;
    async_ld16(dst + lds_off_A8(m, kg), src + (size_t)m * ld + kg * 8);
  }
}
// B tile: 64 cols x 4 kgroups = 256 tasks, 1 per thread.
__device__ __forceinline__ void stage_B_f32(__bf16* dst, const float* __restrict__ src,
                                            int ld, int tid) {
  int n = tid >> 2, kq = tid & 3;
  *(v8bf*)(dst + lds_off_B8(n, kq)) = cvt8(src + (size_t)n * ld + kq * 8);
}
__device__ __forceinline__ void stage_B_bf16(__bf16* dst, const __bf16* __restrict__ src,
                                             int ld, int tid) {
  int n = tid >> 2, kq = tid & 3;
  async_ld16(dst + lds_off_B8(n, kq), src + (size_t)n * ld + kq * 8);
}

// --- WMMA inner product over one staged buffer (4 frag loads, 4 WMMAs) ------
__device__ __forceinline__ void wmma_tile(const __bf16* ldsA, const __bf16* ldsB,
                                          int wm, int wn, int lane, v8f acc[2][2]) {
  v16bf a0 = ((const v16bf*)(ldsA + (wm * 2 + 0) * 512))[lane];
  v16bf a1 = ((const v16bf*)(ldsA + (wm * 2 + 1) * 512))[lane];
  v16bf b0 = ((const v16bf*)(ldsB + (wn * 2 + 0) * 512))[lane];
  v16bf b1 = ((const v16bf*)(ldsB + (wn * 2 + 1) * 512))[lane];
  acc[0][0] = __builtin_amdgcn_wmma_f32_16x16x32_bf16(false, a0, false, b0, (short)0, acc[0][0], false, false);
  acc[0][1] = __builtin_amdgcn_wmma_f32_16x16x32_bf16(false, a0, false, b1, (short)0, acc[0][1], false, false);
  acc[1][0] = __builtin_amdgcn_wmma_f32_16x16x32_bf16(false, a1, false, b0, (short)0, acc[1][0], false, false);
  acc[1][1] = __builtin_amdgcn_wmma_f32_16x16x32_bf16(false, a1, false, b1, (short)0, acc[1][1], false, false);
}

// ---------------------------------------------------------------------------
// Kernel 1: Y = X @ W^T + b (M=8192,N=1024,K=1024), bf16 output.
// vtranspose==0: head-split [B,H,S,DEPTH]; vtranspose==1: [B,H,DEPTH,S].
// ---------------------------------------------------------------------------
__global__ __launch_bounds__(THREADS) void mha_proj_kernel(
    const float* __restrict__ X, const float* __restrict__ W,
    const float* __restrict__ bias, __bf16* __restrict__ Y, int vtranspose) {
  __shared__ __align__(32) __bf16 ldsA[2][BM * BK];
  __shared__ __align__(32) __bf16 ldsB[2][BK * BN];
  const int tid  = threadIdx.x;
  const int lane = tid & 31;
  const int wave = tid >> 5;
  const int wm = wave >> 1, wn = wave & 1;
  const int m0 = blockIdx.x * BM;
  const int n0 = blockIdx.y * BN;
  const float* Abase = X + (size_t)m0 * D_MODEL;
  const float* Bbase = W + (size_t)n0 * D_MODEL;

  v8f acc[2][2] = {};
  stage_A_f32(ldsA[0], Abase, D_MODEL, tid);
  stage_B_f32(ldsB[0], Bbase, D_MODEL, tid);
  const int NK = D_MODEL / BK;  // 32
#pragma unroll 2
  for (int kt = 0; kt < NK; ++kt) {
    __syncthreads();
    if (kt + 1 < NK) {
      stage_A_f32(ldsA[(kt + 1) & 1], Abase + (kt + 1) * BK, D_MODEL, tid);
      stage_B_f32(ldsB[(kt + 1) & 1], Bbase + (kt + 1) * BK, D_MODEL, tid);
      if (kt + 2 < NK)
        __builtin_prefetch(Abase + (size_t)(tid >> 1) * D_MODEL + (kt + 2) * BK, 0, 1);
    }
    wmma_tile(ldsA[kt & 1], ldsB[kt & 1], wm, wn, lane, acc);
  }
#pragma unroll
  for (int fm = 0; fm < 2; ++fm)
#pragma unroll
    for (int fn = 0; fn < 2; ++fn)
#pragma unroll
      for (int r = 0; r < 8; ++r) {
        int m = m0 + wm * 32 + fm * 16 + r + ((lane < 16) ? 0 : 8);
        int n = n0 + wn * 32 + fn * 16 + (lane & 15);
        float v = acc[fm][fn][r] + bias[n];
        int b = m >> 10, s = m & (SEQ - 1);
        int h = n >> 6, d = n & (DEPTH - 1);
        size_t idx = vtranspose
            ? (((size_t)(b * NUM_HEADS + h)) * DEPTH + d) * SEQ + s
            : (((size_t)(b * NUM_HEADS + h)) * SEQ + s) * DEPTH + d;
        Y[idx] = (__bf16)v;
      }
}

// ---------------------------------------------------------------------------
// Kernel 2: logits[bh,s,n] = (Qh[bh] @ Kh[bh]^T)/8 + mask (M=N=1024,K=64).
// Both operands staged with async global->LDS copies.
// ---------------------------------------------------------------------------
__global__ __launch_bounds__(THREADS) void mha_logits_kernel(
    const __bf16* __restrict__ Qh, const __bf16* __restrict__ Kh,
    const float* __restrict__ mask, float* __restrict__ attn) {
  __shared__ __align__(32) __bf16 ldsA[2][BM * BK];
  __shared__ __align__(32) __bf16 ldsB[2][BK * BN];
  const int tid  = threadIdx.x;
  const int lane = tid & 31;
  const int wave = tid >> 5;
  const int wm = wave >> 1, wn = wave & 1;
  const int bh = blockIdx.z;
  const int m0 = blockIdx.x * BM;
  const int n0 = blockIdx.y * BN;
  const __bf16* Abase = Qh + (size_t)bh * SEQ * DEPTH + (size_t)m0 * DEPTH;
  const __bf16* Bbase = Kh + (size_t)bh * SEQ * DEPTH + (size_t)n0 * DEPTH;

  v8f acc[2][2] = {};
  stage_A_bf16(ldsA[0], Abase, DEPTH, tid);
  stage_B_bf16(ldsB[0], Bbase, DEPTH, tid);
  const int NK = DEPTH / BK;  // 2
#pragma unroll
  for (int kt = 0; kt < NK; ++kt) {
    async_wait0();           // our async stages must land before the barrier
    __syncthreads();
    if (kt + 1 < NK) {
      stage_A_bf16(ldsA[(kt + 1) & 1], Abase + (kt + 1) * BK, DEPTH, tid);
      stage_B_bf16(ldsB[(kt + 1) & 1], Bbase + (kt + 1) * BK, DEPTH, tid);
    }
    wmma_tile(ldsA[kt & 1], ldsB[kt & 1], wm, wn, lane, acc);
  }
  const float scale = 0.125f;  // 1/sqrt(64)
#pragma unroll
  for (int fm = 0; fm < 2; ++fm)
#pragma unroll
    for (int fn = 0; fn < 2; ++fn)
#pragma unroll
      for (int r = 0; r < 8; ++r) {
        int s = m0 + wm * 32 + fm * 16 + r + ((lane < 16) ? 0 : 8);
        int n = n0 + wn * 32 + fn * 16 + (lane & 15);
        attn[((size_t)bh * SEQ + s) * SEQ + n] =
            acc[fm][fn][r] * scale + mask[(size_t)s * SEQ + n];
      }
}

// ---------------------------------------------------------------------------
// Kernel 3: in-place row softmax over attn rows (1024 wide). One block/row.
// ---------------------------------------------------------------------------
__global__ __launch_bounds__(THREADS) void mha_softmax_kernel(float* __restrict__ attn) {
  const int tid  = threadIdx.x;
  const int lane = tid & 31;
  const int wave = tid >> 5;
  float* p = attn + (size_t)blockIdx.x * SEQ;
  __shared__ float sred[8];

  float vloc[4];
  float mx = -INFINITY;
#pragma unroll
  for (int j = 0; j < 4; ++j) {
    vloc[j] = p[tid + j * THREADS];
    mx = fmaxf(mx, vloc[j]);
  }
#pragma unroll
  for (int off = 16; off > 0; off >>= 1) mx = fmaxf(mx, __shfl_xor(mx, off, 32));
  if (lane == 0) sred[wave] = mx;
  __syncthreads();
  mx = sred[0];
#pragma unroll
  for (int w = 1; w < 8; ++w) mx = fmaxf(mx, sred[w]);

  float sum = 0.f;
#pragma unroll
  for (int j = 0; j < 4; ++j) {
    vloc[j] = expf(vloc[j] - mx);
    sum += vloc[j];
  }
#pragma unroll
  for (int off = 16; off > 0; off >>= 1) sum += __shfl_xor(sum, off, 32);
  __syncthreads();
  if (lane == 0) sred[wave] = sum;
  __syncthreads();
  sum = 0.f;
#pragma unroll
  for (int w = 0; w < 8; ++w) sum += sred[w];
  float inv = 1.f / sum;
#pragma unroll
  for (int j = 0; j < 4; ++j) p[tid + j * THREADS] = vloc[j] * inv;
}

// ---------------------------------------------------------------------------
// Kernel 4: ctx[b,s,h*64+d] = attn[bh] @ Vh[bh] (M=1024,N=64,K=1024).
// V is stored transposed [B,H,DEPTH,S]; its tile is staged asynchronously.
// ---------------------------------------------------------------------------
__global__ __launch_bounds__(THREADS) void mha_attnv_kernel(
    const float* __restrict__ attn, const __bf16* __restrict__ VhT,
    __bf16* __restrict__ ctx) {
  __shared__ __align__(32) __bf16 ldsA[2][BM * BK];
  __shared__ __align__(32) __bf16 ldsB[2][BK * BN];
  const int tid  = threadIdx.x;
  const int lane = tid & 31;
  const int wave = tid >> 5;
  const int wm = wave >> 1, wn = wave & 1;
  const int bh = blockIdx.z;
  const int m0 = blockIdx.x * BM;          // full N=64 in one block tile
  const int b  = bh >> 4, h = bh & 15;
  const float* Abase = attn + (size_t)bh * SEQ * SEQ + (size_t)m0 * SEQ;
  const __bf16* Bbase = VhT + (size_t)bh * DEPTH * SEQ;  // row d, K-contiguous

  v8f acc[2][2] = {};
  stage_A_f32(ldsA[0], Abase, SEQ, tid);
  stage_B_bf16(ldsB[0], Bbase, SEQ, tid);
  const int NK = SEQ / BK;  // 32
#pragma unroll 2
  for (int kt = 0; kt < NK; ++kt) {
    async_wait0();
    __syncthreads();
    if (kt + 1 < NK) {
      stage_A_f32(ldsA[(kt + 1) & 1], Abase + (kt + 1) * BK, SEQ, tid);
      stage_B_bf16(ldsB[(kt + 1) & 1], Bbase + (kt + 1) * BK, SEQ, tid);
      if (kt + 2 < NK)
        __builtin_prefetch(Abase + (size_t)(tid >> 1) * SEQ + (kt + 2) * BK, 0, 1);
    }
    wmma_tile(ldsA[kt & 1], ldsB[kt & 1], wm, wn, lane, acc);
  }
#pragma unroll
  for (int fm = 0; fm < 2; ++fm)
#pragma unroll
    for (int fn = 0; fn < 2; ++fn)
#pragma unroll
      for (int r = 0; r < 8; ++r) {
        int s = m0 + wm * 32 + fm * 16 + r + ((lane < 16) ? 0 : 8);
        int d = wn * 32 + fn * 16 + (lane & 15);
        ctx[((size_t)b * SEQ + s) * D_MODEL + h * DEPTH + d] = (__bf16)acc[fm][fn][r];
      }
}

// ---------------------------------------------------------------------------
// Kernel 5: out = ctx @ fc_w^T + fc_b (M=8192,N=1024,K=1024), fp32 out.
// ctx tile staged asynchronously; weight tile converted fp32 -> bf16.
// ---------------------------------------------------------------------------
__global__ __launch_bounds__(THREADS) void mha_fc_kernel(
    const __bf16* __restrict__ ctx, const float* __restrict__ W,
    const float* __restrict__ bias, float* __restrict__ out) {
  __shared__ __align__(32) __bf16 ldsA[2][BM * BK];
  __shared__ __align__(32) __bf16 ldsB[2][BK * BN];
  const int tid  = threadIdx.x;
  const int lane = tid & 31;
  const int wave = tid >> 5;
  const int wm = wave >> 1, wn = wave & 1;
  const int m0 = blockIdx.x * BM;
  const int n0 = blockIdx.y * BN;
  const __bf16* Abase = ctx + (size_t)m0 * D_MODEL;
  const float*  Bbase = W + (size_t)n0 * D_MODEL;

  v8f acc[2][2] = {};
  stage_A_bf16(ldsA[0], Abase, D_MODEL, tid);
  stage_B_f32(ldsB[0], Bbase, D_MODEL, tid);
  const int NK = D_MODEL / BK;  // 32
#pragma unroll 2
  for (int kt = 0; kt < NK; ++kt) {
    async_wait0();
    __syncthreads();
    if (kt + 1 < NK) {
      stage_A_bf16(ldsA[(kt + 1) & 1], Abase + (kt + 1) * BK, D_MODEL, tid);
      stage_B_f32(ldsB[(kt + 1) & 1], Bbase + (kt + 1) * BK, D_MODEL, tid);
      if (kt + 2 < NK)
        __builtin_prefetch(Bbase + (size_t)(tid >> 1) * D_MODEL + (kt + 2) * BK, 0, 1);
    }
    wmma_tile(ldsA[kt & 1], ldsB[kt & 1], wm, wn, lane, acc);
  }
#pragma unroll
  for (int fm = 0; fm < 2; ++fm)
#pragma unroll
    for (int fn = 0; fn < 2; ++fn)
#pragma unroll
      for (int r = 0; r < 8; ++r) {
        int m = m0 + wm * 32 + fm * 16 + r + ((lane < 16) ? 0 : 8);
        int n = n0 + wn * 32 + fn * 16 + (lane & 15);
        out[(size_t)m * D_MODEL + n] = acc[fm][fn][r] + bias[n];
      }
}

// ---------------------------------------------------------------------------
extern "C" void kernel_launch(void* const* d_in, const int* in_sizes, int n_in,
                              void* d_out, int out_size, void* d_ws, size_t ws_size,
                              hipStream_t stream) {
  const float* q    = (const float*)d_in[0];
  const float* k    = (const float*)d_in[1];
  const float* v    = (const float*)d_in[2];
  const float* mask = (const float*)d_in[3];
  const float* wq_w = (const float*)d_in[4];
  const float* wq_b = (const float*)d_in[5];
  const float* wk_w = (const float*)d_in[6];
  const float* wk_b = (const float*)d_in[7];
  const float* wv_w = (const float*)d_in[8];
  const float* wv_b = (const float*)d_in[9];
  const float* fc_w = (const float*)d_in[10];
  const float* fc_b = (const float*)d_in[11];

  float* out  = (float*)d_out;                  // [B, S, D]
  float* attn = out + (size_t)BS * D_MODEL;     // [B, H, S, S]

  // workspace: qh | kh | vT | ctx  (each BS*D bf16 = 16 MiB; total 64 MiB)
  __bf16* qh  = (__bf16*)d_ws;
  __bf16* kh  = qh + (size_t)BS * D_MODEL;
  __bf16* vt  = kh + (size_t)BS * D_MODEL;
  __bf16* ctx = vt + (size_t)BS * D_MODEL;

  dim3 blk(THREADS);
  dim3 gproj(BS / BM, D_MODEL / BN);            // 64 x 16
  mha_proj_kernel<<<gproj, blk, 0, stream>>>(q, wq_w, wq_b, qh, 0);
  mha_proj_kernel<<<gproj, blk, 0, stream>>>(k, wk_w, wk_b, kh, 0);
  mha_proj_kernel<<<gproj, blk, 0, stream>>>(v, wv_w, wv_b, vt, 1);

  dim3 glog(SEQ / BM, SEQ / BN, BH);            // 8 x 16 x 128
  mha_logits_kernel<<<glog, blk, 0, stream>>>(qh, kh, mask, attn);

  mha_softmax_kernel<<<dim3(BH * SEQ), blk, 0, stream>>>(attn);

  dim3 gav(SEQ / BM, 1, BH);                    // 8 x 1 x 128
  mha_attnv_kernel<<<gav, blk, 0, stream>>>(attn, vt, ctx);

  mha_fc_kernel<<<gproj, blk, 0, stream>>>(ctx, fc_w, fc_b, out);
}
